// RainfallLSTM_39453569581517
// MI455X (gfx1250) — matrix operand order repaired and checked by reading, not compiled, not run on hardware
//
#include <hip/hip_runtime.h>

typedef __attribute__((ext_vector_type(16))) _Float16 v16h;
typedef __attribute__((ext_vector_type(8)))  float    v8f;

#define B_TOT 32768
#define T_SEQ 7
#define D_IN  40
#define HDIM  64
#define G4    256      // 4*H
#define FCW   32

#define WAVES_PER_WG   8
#define ROWS_PER_WAVE  16
#define ROWS_PER_WG    (WAVES_PER_WG * ROWS_PER_WAVE)   // 128

#define WMMA_F16(A, Bf, C) \
  __builtin_amdgcn_wmma_f32_16x16x32_f16(false, (A), false, (Bf), (short)0, (C), false, false)

__device__ __forceinline__ float sigmoid_f(float v) {
  return 1.0f / (1.0f + __expf(-v));
}

// Branch-free tanh: 1 - 2/(e^{2x}+1). Saturates to +/-1, no NaN for |x| large.
__device__ __forceinline__ float tanh_f(float v) {
  float e = __expf(2.0f * v);
  return 1.0f - 2.0f / (e + 1.0f);
}

// ---- h buffers are f16, stored in "A-fragment-permuted" column order ----
// A-frag (16x32 f16) element e of lane (M=lane&15, c=lane>>4) holds
//   K = k0 + (e>=8?16:0) + (e&7) + 8*c
// We store h[M][p] with p = K with bits 3 and 4 swapped, so a whole fragment
// is 16 contiguous f16 per lane: offset M*64 + f*32 + c*16  (f = k0/32).
__device__ __forceinline__ int perm_col(int n) {
  return (n & ~24) | ((n & 8) << 1) | ((n & 16) >> 1);
}

__device__ __forceinline__ v16h load_a_frag_h(const _Float16* buf, int lane, int f) {
  const int M = lane & 15;
  const int c = lane >> 4;
  return *(const v16h*)(buf + M * HDIM + f * 32 + c * 16);
}

// A fragment for the x input (K padded 40 -> 64 with zeros), straight from global fp32.
__device__ __forceinline__ v16h load_a_frag_x(const float* __restrict__ xw, int lane, int t, int k0) {
  const int M = lane & 15;
  const int c = lane >> 4;
  const float* row = xw + (size_t)M * (T_SEQ * D_IN) + t * D_IN;
  v16h a;
#pragma unroll
  for (int v = 0; v < 8; ++v) {
    int kb = k0 + ((v & 4) << 2) + ((v & 3) << 1) + (c << 3);
    float f0 = 0.0f, f1 = 0.0f;
    if (kb < D_IN) { f0 = row[kb]; f1 = row[kb + 1]; }   // D_IN even -> kb+1 also in range
    a[2 * v]     = (_Float16)f0;
    a[2 * v + 1] = (_Float16)f1;
  }
  return a;
}

// B fragment (32x16 f16) from f16 LDS weights stored [N][64] row-major:
// lane N = n0 + (lane&15); K = k0 + 8*(lane>>4) + {0..7} and k0 + 16 + 8*(lane>>4) + {0..7}.
// Plain (non-volatile) load -> pipelined ds_load_b128 pair; hoisting out of the
// time loop is prevented by the per-iteration memory clobber, and bulk
// prefetching is limited by sched_barrier(0) per gate tile.
__device__ __forceinline__ v16h load_b_frag(const _Float16* w, int lane, int n0, int k0) {
  typedef __attribute__((ext_vector_type(8))) _Float16 v8h_t;
  const int n  = n0 + (lane & 15);
  const int kb = k0 + ((lane >> 4) << 3);
  v8h_t lo = *(const v8h_t*)(w + n * HDIM + kb);
  v8h_t hi = *(const v8h_t*)(w + n * HDIM + kb + 16);
  return __builtin_shufflevector(lo, hi, 0,1,2,3,4,5,6,7,8,9,10,11,12,13,14,15);
}

// One LSTM step for a 16-row tile. c[4] = persistent fp32 cell tiles (D layout).
// hbuf = [16][64] f16 LDS (permuted cols), overwritten with the new h.
__device__ __forceinline__ void lstm_step(
    v16h ax0, v16h ax1, v16h ah0, v16h ah1,
    const _Float16* Wi, const _Float16* Wh, const float* bias,
    v8f c[4], _Float16* hbuf, int lane)
{
  const int M0 = lane & 15;
  const int cc = lane >> 4;
#pragma unroll
  for (int j = 0; j < 4; ++j) {           // 16-column group within H
    v8f acc[4];                           // i, f, g, o tiles
#pragma unroll
    for (int g = 0; g < 4; ++g) {
      const int n0 = g * 64 + j * 16;     // gate order i,f,g,o
      const float bv = bias[n0 + M0];
      v8f a;
#pragma unroll
      for (int r = 0; r < 8; ++r) a[r] = bv;
      a = WMMA_F16(ax0, load_b_frag(Wi, lane, n0,  0), a);
      a = WMMA_F16(ax1, load_b_frag(Wi, lane, n0, 32), a);
      a = WMMA_F16(ah0, load_b_frag(Wh, lane, n0,  0), a);
      a = WMMA_F16(ah1, load_b_frag(Wh, lane, n0, 32), a);
      acc[g] = a;
      __builtin_amdgcn_sched_barrier(0);  // keep B-frag loads local to this tile
    }
    const int pcol = perm_col(j * 16 + M0);
#pragma unroll
    for (int r = 0; r < 8; ++r) {
      float ig = sigmoid_f(acc[0][r]);
      float fg = sigmoid_f(acc[1][r]);
      float gg = tanh_f(acc[2][r]);
      float og = sigmoid_f(acc[3][r]);
      float cv = fg * c[j][r] + ig * gg;
      c[j][r] = cv;
      hbuf[(r + (cc << 3)) * HDIM + pcol] = (_Float16)(og * tanh_f(cv));
    }
  }
}

extern "C" __global__ __launch_bounds__(256)
void rainfall_lstm_fused(
    const float* __restrict__ x,
    const float* __restrict__ Wih0, const float* __restrict__ Whh0,
    const float* __restrict__ bih0, const float* __restrict__ bhh0,
    const float* __restrict__ Wih1, const float* __restrict__ Whh1,
    const float* __restrict__ bih1, const float* __restrict__ bhh1,
    const float* __restrict__ W1,   const float* __restrict__ b1,
    const float* __restrict__ W2,   const float* __restrict__ b2,
    float* __restrict__ out)
{
  __shared__ __attribute__((aligned(32))) _Float16 sWih0[G4][HDIM];  // K padded 40->64
  __shared__ __attribute__((aligned(32))) _Float16 sWhh0[G4][HDIM];
  __shared__ __attribute__((aligned(32))) _Float16 sWih1[G4][HDIM];
  __shared__ __attribute__((aligned(32))) _Float16 sWhh1[G4][HDIM];
  __shared__ __attribute__((aligned(32))) _Float16 sW1[FCW][HDIM];
  __shared__ float sB0[G4];
  __shared__ float sB1[G4];
  __shared__ float sBfc[FCW];
  __shared__ float sW2[FCW];
  __shared__ __attribute__((aligned(32))) _Float16 sH1[WAVES_PER_WG][ROWS_PER_WAVE][HDIM];
  __shared__ __attribute__((aligned(32))) _Float16 sH2[WAVES_PER_WG][ROWS_PER_WAVE][HDIM];
  __shared__ float sHid[WAVES_PER_WG][ROWS_PER_WAVE][FCW];

  const int tid = threadIdx.x;

  // ---- cooperative weight load + f16 convert ----
  for (int idx = tid; idx < G4 * HDIM; idx += 256) {
    const int n = idx >> 6, k = idx & 63;
    sWih0[n][k] = (_Float16)((k < D_IN) ? Wih0[n * D_IN + k] : 0.0f);
    sWhh0[n][k] = (_Float16)Whh0[idx];
    sWih1[n][k] = (_Float16)Wih1[idx];
    sWhh1[n][k] = (_Float16)Whh1[idx];
  }
  for (int idx = tid; idx < FCW * HDIM; idx += 256)
    sW1[idx >> 6][idx & 63] = (_Float16)W1[idx];
  for (int idx = tid; idx < G4; idx += 256) {
    sB0[idx] = bih0[idx] + bhh0[idx];
    sB1[idx] = bih1[idx] + bhh1[idx];
  }
  if (tid < FCW) { sBfc[tid] = b1[tid]; sW2[tid] = W2[tid]; }
  __syncthreads();

  const int wave = tid >> 5;
  const int lane = tid & 31;
  const int m0   = blockIdx.x * ROWS_PER_WG + wave * ROWS_PER_WAVE;

  _Float16* h1 = &sH1[wave][0][0];
  _Float16* h2 = &sH2[wave][0][0];
  for (int i = lane; i < ROWS_PER_WAVE * HDIM; i += 32) {
    h1[i] = (_Float16)0.0f;
    h2[i] = (_Float16)0.0f;
  }

  v8f c1[4], c2[4];
#pragma unroll
  for (int j = 0; j < 4; ++j) { c1[j] = (v8f){}; c2[j] = (v8f){}; }

  const float* xw = x + (size_t)m0 * (T_SEQ * D_IN);

#pragma clang loop unroll(disable)
  for (int t = 0; t < T_SEQ; ++t) {
    asm volatile("" ::: "memory");   // stop cross-iteration hoisting of LDS reads

    // layer 0: input = x_t (K padded to 64), recurrent = h1
    v16h ax0 = load_a_frag_x(xw, lane, t, 0);
    v16h ax1 = load_a_frag_x(xw, lane, t, 32);
    v16h ah0 = load_a_frag_h(h1, lane, 0);
    v16h ah1 = load_a_frag_h(h1, lane, 1);
    lstm_step(ax0, ax1, ah0, ah1, &sWih0[0][0], &sWhh0[0][0], sB0, c1, h1, lane);

    // layer 1: input = new h1, recurrent = h2
    v16h a10 = load_a_frag_h(h1, lane, 0);
    v16h a11 = load_a_frag_h(h1, lane, 1);
    v16h a20 = load_a_frag_h(h2, lane, 0);
    v16h a21 = load_a_frag_h(h2, lane, 1);
    lstm_step(a10, a11, a20, a21, &sWih1[0][0], &sWhh1[0][0], sB1, c2, h2, lane);
  }

  // ---- FC head: hid = relu(h2 @ W1^T + b1) ----
  v16h f0 = load_a_frag_h(h2, lane, 0);
  v16h f1 = load_a_frag_h(h2, lane, 1);
  const int M0 = lane & 15;
  const int cc = lane >> 4;
  float* hid = &sHid[wave][0][0];
#pragma unroll
  for (int n = 0; n < 2; ++n) {
    const float bv = sBfc[n * 16 + M0];
    v8f acc;
#pragma unroll
    for (int r = 0; r < 8; ++r) acc[r] = bv;
    acc = WMMA_F16(f0, load_b_frag(&sW1[0][0], lane, n * 16,  0), acc);
    acc = WMMA_F16(f1, load_b_frag(&sW1[0][0], lane, n * 16, 32), acc);
#pragma unroll
    for (int r = 0; r < 8; ++r)
      hid[(r + (cc << 3)) * FCW + n * 16 + M0] = fmaxf(acc[r], 0.0f);
  }

  // ---- out = hid @ W2^T + b2 ----
  if (lane < 16) {
    float s = b2[0];
#pragma unroll
    for (int k = 0; k < FCW; ++k) s += hid[lane * FCW + k] * sW2[k];
    out[m0 + lane] = s;
  }
}

extern "C" void kernel_launch(void* const* d_in, const int* in_sizes, int n_in,
                              void* d_out, int out_size, void* d_ws, size_t ws_size,
                              hipStream_t stream) {
  const float* x    = (const float*)d_in[0];
  const float* Wih0 = (const float*)d_in[1];
  const float* Whh0 = (const float*)d_in[2];
  const float* bih0 = (const float*)d_in[3];
  const float* bhh0 = (const float*)d_in[4];
  const float* Wih1 = (const float*)d_in[5];
  const float* Whh1 = (const float*)d_in[6];
  const float* bih1 = (const float*)d_in[7];
  const float* bhh1 = (const float*)d_in[8];
  const float* W1   = (const float*)d_in[9];
  const float* b1   = (const float*)d_in[10];
  const float* W2   = (const float*)d_in[11];
  const float* b2   = (const float*)d_in[12];

  dim3 grid(B_TOT / ROWS_PER_WG);   // 256 workgroups
  dim3 block(256);                  // 8 waves of 32
  hipLaunchKernelGGL(rainfall_lstm_fused, grid, block, 0, stream,
                     x, Wih0, Whh0, bih0, bhh0, Wih1, Whh1, bih1, bhh1,
                     W1, b1, W2, b2, (float*)d_out);
}
